// MultiHeadAttention_72877005078903
// MI455X (gfx1250) — compile-verified, matrix operand order checked by reading, not run on hardware
//
#include <hip/hip_runtime.h>

#define BB  2
#define TT  2048
#define DDIM 1024
#define NHH 16
#define HDD 64

typedef __attribute__((ext_vector_type(16))) __bf16 v16bf;
typedef __attribute__((ext_vector_type(8)))  float  v8f;
typedef __attribute__((ext_vector_type(4)))  int    vi4;

#if defined(__gfx1250__) && __has_builtin(__builtin_amdgcn_global_load_async_to_lds_b128) && \
    __has_builtin(__builtin_amdgcn_s_wait_asynccnt)
#define HAVE_ASYNC_LDS 1
#define GLOBAL_AS __attribute__((address_space(1)))
#define LDS_AS    __attribute__((address_space(3)))
#else
#define HAVE_ASYNC_LDS 0
#endif

// Async 16B global->LDS transfer (no VGPR round trip), else sync fallback.
__device__ __forceinline__ void copy16_to_lds(void* lds_dst, const void* gsrc) {
#if HAVE_ASYNC_LDS
  __builtin_amdgcn_global_load_async_to_lds_b128(
      (GLOBAL_AS vi4*)gsrc, (LDS_AS vi4*)lds_dst, 0, 0);
#else
  *(uint4*)lds_dst = *(const uint4*)gsrc;
#endif
}
__device__ __forceinline__ void wait_lds_copies() {
#if HAVE_ASYNC_LDS
  __builtin_amdgcn_s_wait_asynccnt(0);
#endif
}

// Load a 16x32 bf16 A- or B-fragment from LDS per the CDNA5 wave32 layout:
// lane L holds row (base + (L&15)); VGPRs 0-3 = K {hi*8 .. hi*8+7},
// VGPRs 4-7 = K {16+hi*8 .. 16+hi*8+7}, hi = L>>4.  Both 8-element runs are
// 16B aligned -> two ds_load_b128 per fragment.
__device__ __forceinline__ v16bf load_frag(const __bf16* p, int stride, int row,
                                           int kk, int hi) {
  union { v16bf v; uint4 q[2]; } f;
  const __bf16* r = p + row * stride + kk + hi * 8;
  f.q[0] = *(const uint4*)(r);
  f.q[1] = *(const uint4*)(r + 16);
  return f.v;
}

__device__ __forceinline__ float rmax16(float v) {
  for (int m = 1; m < 16; m <<= 1) v = fmaxf(v, __shfl_xor(v, m, 32));
  return v;
}
__device__ __forceinline__ float rsum16(float v) {
  for (int m = 1; m < 16; m <<= 1) v += __shfl_xor(v, m, 32);
  return v;
}

__global__ void cvt_f32_bf16(const float* __restrict__ src,
                             __bf16* __restrict__ dst, int n4) {
  int i = blockIdx.x * blockDim.x + threadIdx.x;
  if (i < n4) {
    float4 v = ((const float4*)src)[i];
    union { __bf16 h[4]; uint2 u; } o;
    o.h[0] = (__bf16)v.x; o.h[1] = (__bf16)v.y;
    o.h[2] = (__bf16)v.z; o.h[3] = (__bf16)v.w;
    ((uint2*)dst)[i] = o.u;
  }
}

// C[M,N] = A[M,K] * W[N,K]^T, bf16 inputs, f32 accumulate.
// Block tile 128x64, BK=32, 8 waves; each wave owns a 32x32 output
// (2 A-frags x 2 B-frags -> 4 WMMAs per k-step, 2 ds_load_b128 per WMMA).
// mode 0: out = bf16 scattered into [B,NH,T,hd] layout, val=(acc+bias)*scale
// mode 1: out = f32 row-major [M,N],                   val=(acc+bias)*scale
__global__ __launch_bounds__(256) void gemm_bf16(
    const __bf16* __restrict__ A, const __bf16* __restrict__ W,
    const float* __restrict__ bias, float scale, void* __restrict__ out,
    int M, int N, int K, int mode) {
  __shared__ __align__(16) __bf16 As[128 * 32];   // 8 KB
  __shared__ __align__(16) __bf16 Ws[64 * 32];    // 4 KB
  const int tid  = threadIdx.x;
  const int wave = tid >> 5, lane = tid & 31, ln = lane & 15, hi = lane >> 4;
  const int m0 = blockIdx.y * 128, n0 = blockIdx.x * 64;
  const int mrow = (wave >> 1) * 32;   // wave's 32-row m sub-block
  const int ncol = (wave & 1) * 32;    // wave's 32-col n sub-block
  const int lr = (tid * 8) / 32;       // cooperative-load row (0..63)
  const int lc = (tid * 8) % 32;       // cooperative-load col (16B aligned)
  v8f acc00 = {}, acc01 = {}, acc10 = {}, acc11 = {};
  for (int k0 = 0; k0 < K; k0 += 32) {
    __syncthreads();
    copy16_to_lds(As + lr * 32 + lc,        A + (size_t)(m0 + lr) * K + k0 + lc);
    copy16_to_lds(As + (lr + 64) * 32 + lc, A + (size_t)(m0 + lr + 64) * K + k0 + lc);
    copy16_to_lds(Ws + lr * 32 + lc,        W + (size_t)(n0 + lr) * K + k0 + lc);
    if (k0 + 32 < K) {  // prefetch next k-chunk -> global_prefetch_b8
      __builtin_prefetch(A + (size_t)(m0 + lr) * K + k0 + 32 + lc, 0, 1);
      __builtin_prefetch(W + (size_t)(n0 + lr) * K + k0 + 32 + lc, 0, 1);
    }
    wait_lds_copies();
    __syncthreads();
    v16bf a0 = load_frag(As, 32, mrow + ln, 0, hi);
    v16bf a1 = load_frag(As, 32, mrow + 16 + ln, 0, hi);
    v16bf b0 = load_frag(Ws, 32, ncol + ln, 0, hi);
    v16bf b1 = load_frag(Ws, 32, ncol + 16 + ln, 0, hi);
    acc00 = __builtin_amdgcn_wmma_f32_16x16x32_bf16(false, a0, false, b0,
                                                    (short)0, acc00, false, false);
    acc01 = __builtin_amdgcn_wmma_f32_16x16x32_bf16(false, a0, false, b1,
                                                    (short)0, acc01, false, false);
    acc10 = __builtin_amdgcn_wmma_f32_16x16x32_bf16(false, a1, false, b0,
                                                    (short)0, acc10, false, false);
    acc11 = __builtin_amdgcn_wmma_f32_16x16x32_bf16(false, a1, false, b1,
                                                    (short)0, acc11, false, false);
  }
  v8f accs[2][2] = {{acc00, acc01}, {acc10, acc11}};
  for (int mi = 0; mi < 2; ++mi)
    for (int ni = 0; ni < 2; ++ni) {
      v8f acc = accs[mi][ni];
      const int n = n0 + ncol + ni * 16 + ln;
      const float bv = bias[n];
      for (int j = 0; j < 8; ++j) {
        const int m = m0 + mrow + mi * 16 + j + 8 * hi;  // C: VGPR j -> M=j+8*hi
        const float val = (acc[j] + bv) * scale;
        if (mode == 0) {
          const int b = m / TT, trow = m % TT, h = n / HDD, c = n % HDD;
          ((__bf16*)out)[(((size_t)(b * NHH + h) * TT + trow) * HDD) + c] =
              (__bf16)val;
        } else {
          ((float*)out)[(size_t)m * N + n] = val;
        }
      }
    }
}

// Flash attention: grid = B*NH*(T/64) blocks, 128 threads (4 waves).
// Wave w owns query rows [i0+16w, i0+16w+16); streams 64-key tiles causally.
__global__ __launch_bounds__(128) void flash_attn(
    const __bf16* __restrict__ Q, const __bf16* __restrict__ Kh,
    const __bf16* __restrict__ V, __bf16* __restrict__ O) {
  __shared__ __align__(16) __bf16 Qs[64 * 64];
  __shared__ __align__(16) __bf16 Ks[64 * 64];
  __shared__ __align__(16) __bf16 Vt[64 * 64];   // transposed: Vt[hd][key]
  __shared__ __align__(16) __bf16 Ps[64 * 64];
  const int tid  = threadIdx.x;
  const int wave = tid >> 5, lane = tid & 31, ln = lane & 15, hi = lane >> 4;
  const int ntq = TT / 64;
  int bi = blockIdx.x;
  const int qt = bi % ntq; bi /= ntq;
  const int h  = bi % NHH;
  const int b  = bi / NHH;
  const int i0 = qt * 64;
  const float slope = exp2f(-0.5f * (float)(h + 1));  // 2^(-8*(h+1)/16)
  const size_t headbase = ((size_t)(b * NHH + h)) * TT * HDD;

  {
    const __bf16* gq = Q + headbase + (size_t)i0 * HDD;
    for (int it = 0; it < 4; ++it)
      copy16_to_lds(((uint4*)Qs) + tid + it * 128,
                    ((const uint4*)gq) + tid + it * 128);
    wait_lds_copies();
  }

  float rowmax[8], rowsum[8];
  v8f oacc[4];
  const v8f vz = {};
  for (int j = 0; j < 8; ++j) { rowmax[j] = -3.0e38f; rowsum[j] = 0.f; }
  for (int od = 0; od < 4; ++od) oacc[od] = vz;

  for (int kt = 0; kt <= qt; ++kt) {          // causal: skip masked tiles
    const int j0 = kt * 64;
    __syncthreads();
    const __bf16* gk = Kh + headbase + (size_t)j0 * HDD;
    const __bf16* gv = V  + headbase + (size_t)j0 * HDD;
    for (int it = 0; it < 4; ++it) {
      copy16_to_lds(((uint4*)Ks) + tid + it * 128,
                    ((const uint4*)gk) + tid + it * 128);
      // V transposed into LDS so P@V B-fragments are contiguous b128 loads
      const int e  = (tid + it * 128) * 8;
      const int vr = e / 64, vc = e % 64;
      union { uint4 u; __bf16 hh[8]; } vld;
      vld.u = ((const uint4*)gv)[tid + it * 128];
      for (int j = 0; j < 8; ++j) Vt[(vc + j) * 64 + vr] = vld.hh[j];
    }
    wait_lds_copies();
    __syncthreads();

    // S = Q * K^T  (4 n-tiles x 2 k-steps of WMMA)
    v8f S[4];
    for (int nt = 0; nt < 4; ++nt) {
      v8f s = {};
      for (int kk = 0; kk < HDD; kk += 32) {
        v16bf a  = load_frag(Qs, 64, wave * 16 + ln, kk, hi);
        v16bf bb = load_frag(Ks, 64, nt * 16 + ln, kk, hi);
        s = __builtin_amdgcn_wmma_f32_16x16x32_bf16(false, a, false, bb,
                                                    (short)0, s, false, false);
      }
      S[nt] = s;
    }

    // ALiBi + causal mask, in accumulator layout
    for (int nt = 0; nt < 4; ++nt) {
      const int gj = j0 + nt * 16 + ln;
      for (int j = 0; j < 8; ++j) {
        const int gi = i0 + wave * 16 + j + 8 * hi;
        float s = S[nt][j] + slope * (float)(gj - gi);
        S[nt][j] = (gj > gi) ? -3.0e38f : s;
      }
    }

    // online softmax
    float newmax[8];
    for (int j = 0; j < 8; ++j) newmax[j] = rowmax[j];
    for (int nt = 0; nt < 4; ++nt)
      for (int j = 0; j < 8; ++j)
        newmax[j] = fmaxf(newmax[j], rmax16(S[nt][j]));

    float fs[8], psum[8];
    for (int j = 0; j < 8; ++j) {
      fs[j] = __expf(rowmax[j] - newmax[j]);
      rowmax[j] = newmax[j];
      psum[j] = 0.f;
    }
    for (int nt = 0; nt < 4; ++nt)
      for (int j = 0; j < 8; ++j) {
        const float p = __expf(S[nt][j] - rowmax[j]);
        psum[j] += p;
        Ps[(wave * 16 + j + 8 * hi) * 64 + nt * 16 + ln] = (__bf16)p;
      }
    for (int j = 0; j < 8; ++j)
      rowsum[j] = rowsum[j] * fs[j] + rsum16(psum[j]);
    for (int od = 0; od < 4; ++od)
      for (int j = 0; j < 8; ++j) oacc[od][j] *= fs[j];

    // O += P * V (Ps slab is wave-private; same-wave LDS ops are in order)
    for (int od = 0; od < 4; ++od)
      for (int kk = 0; kk < 64; kk += 32) {
        v16bf a  = load_frag(Ps + wave * 16 * 64, 64, ln, kk, hi);
        v16bf bb = load_frag(Vt, 64, od * 16 + ln, kk, hi);
        oacc[od] = __builtin_amdgcn_wmma_f32_16x16x32_bf16(
            false, a, false, bb, (short)0, oacc[od], false, false);
      }
  }

  // normalize + merge heads into [B,T,D] bf16
  for (int j = 0; j < 8; ++j) {
    const int t = i0 + wave * 16 + j + 8 * hi;
    const float inv = 1.0f / rowsum[j];
    for (int od = 0; od < 4; ++od) {
      const int c = h * HDD + od * 16 + ln;
      O[((size_t)(b * TT + t)) * DDIM + c] = (__bf16)(oacc[od][j] * inv);
    }
  }
}

extern "C" void kernel_launch(void* const* d_in, const int* in_sizes, int n_in,
                              void* d_out, int out_size, void* d_ws, size_t ws_size,
                              hipStream_t stream) {
  (void)in_sizes; (void)n_in; (void)out_size; (void)ws_size;
  const float* x  = (const float*)d_in[0];
  const float* Wq = (const float*)d_in[1];
  const float* bq = (const float*)d_in[2];
  const float* Wk = (const float*)d_in[3];
  const float* bk = (const float*)d_in[4];
  const float* Wv = (const float*)d_in[5];
  const float* bv = (const float*)d_in[6];
  const float* Wp = (const float*)d_in[7];
  const float* bp = (const float*)d_in[8];
  float* out = (float*)d_out;

  const size_t nx = (size_t)BB * TT * DDIM;   // 4M elems
  const size_t nw = (size_t)DDIM * DDIM;      // 1M elems
  char* p = (char*)d_ws;
  __bf16* xb  = (__bf16*)p; p += nx * 2;
  __bf16* wqb = (__bf16*)p; p += nw * 2;
  __bf16* wkb = (__bf16*)p; p += nw * 2;
  __bf16* wvb = (__bf16*)p; p += nw * 2;
  __bf16* wpb = (__bf16*)p; p += nw * 2;
  __bf16* Qh  = (__bf16*)p; p += nx * 2;
  __bf16* Kb  = (__bf16*)p; p += nx * 2;
  __bf16* Vb  = (__bf16*)p; p += nx * 2;
  __bf16* Ob  = (__bf16*)p; p += nx * 2;

  const int M = BB * TT;   // 4096
  const float qkscale = 0.17677669529663687f;  // 1024^-0.25

  cvt_f32_bf16<<<dim3((unsigned)(nx / 4 / 256)), 256, 0, stream>>>(x,  xb,  (int)(nx / 4));
  cvt_f32_bf16<<<dim3((unsigned)(nw / 4 / 256)), 256, 0, stream>>>(Wq, wqb, (int)(nw / 4));
  cvt_f32_bf16<<<dim3((unsigned)(nw / 4 / 256)), 256, 0, stream>>>(Wk, wkb, (int)(nw / 4));
  cvt_f32_bf16<<<dim3((unsigned)(nw / 4 / 256)), 256, 0, stream>>>(Wv, wvb, (int)(nw / 4));
  cvt_f32_bf16<<<dim3((unsigned)(nw / 4 / 256)), 256, 0, stream>>>(Wp, wpb, (int)(nw / 4));

  dim3 ggrid(DDIM / 64, M / 128);   // (16, 32)
  gemm_bf16<<<ggrid, 256, 0, stream>>>(xb, wqb, bq, qkscale, Qh, M, DDIM, DDIM, 0);
  gemm_bf16<<<ggrid, 256, 0, stream>>>(xb, wkb, bk, qkscale, Kb, M, DDIM, DDIM, 0);
  gemm_bf16<<<ggrid, 256, 0, stream>>>(xb, wvb, bv, 1.0f,    Vb, M, DDIM, DDIM, 0);

  flash_attn<<<dim3(BB * NHH * (TT / 64)), 128, 0, stream>>>(Qh, Kb, Vb, Ob);

  gemm_bf16<<<ggrid, 256, 0, stream>>>(Ob, wpb, bp, 1.0f, out, M, DDIM, DDIM, 1);
}
